// EGNN_20701742367343
// MI455X (gfx1250) — matrix operand order repaired and checked by reading, not compiled, not run on hardware
//
#include <hip/hip_runtime.h>
#include <hip/hip_bf16.h>

#define N_NODES 20000
#define N_EDGES 320000
#define HDIM    128
#define NLAYERS 4
#define KE1_VALID 289      /* 2H + 1 + D_EDGE */
#define KT_E1   10         /* 320 / 32 */
#define KT_E2   4          /* 128 / 32 */
#define KT_N1   8          /* 256 / 32 */
#define KT_N2   4
#define NTILES  8          /* 128 / 16 */
#define PAD     136        /* bf16 row stride for LDS transpose slab */

#define GLOBAL_AS __attribute__((address_space(1)))
#define LDS_AS    __attribute__((address_space(3)))

typedef __attribute__((ext_vector_type(16))) __bf16        v16bf;
typedef __attribute__((ext_vector_type(8)))  float         v8f;
typedef __attribute__((ext_vector_type(16))) unsigned short v16us;
typedef __attribute__((ext_vector_type(8)))  unsigned short v8us;
typedef __attribute__((ext_vector_type(4)))  int           v4i;

union Frag {
    v16us us;
    v16bf bf;
    unsigned short s[16];
};

__device__ __forceinline__ unsigned short f2bf(float f) {
    unsigned int u = __builtin_bit_cast(unsigned int, f);
    u += 0x7FFFu + ((u >> 16) & 1u);          // round-to-nearest-even
    return (unsigned short)(u >> 16);
}

__device__ __forceinline__ float silu_f(float x) {
    return x / (1.0f + __expf(-x));
}

__device__ __forceinline__ v8f wmma_bf16(const Frag& a, const Frag& b, v8f c) {
    return __builtin_amdgcn_wmma_f32_16x16x32_bf16(
        false, a.bf, false, b.bf, (short)0, c, false, false);
}

// A-fragment (16-bit A 16x32 ISA layout) from 32 consecutive bf16 of one row.
// Lane half `hi` owns K runs [8hi, 8hi+8) and [16+8hi, 16+8hi+8).
__device__ __forceinline__ Frag load_frag_bf16(const unsigned short* __restrict__ base, int hi) {
    Frag a;
    v8us lo = *(const v8us*)(base + 8 * hi);
    v8us hh = *(const v8us*)(base + 16 + 8 * hi);
#pragma unroll
    for (int j = 0; j < 8; ++j) { a.s[j] = lo[j]; a.s[8 + j] = hh[j]; }
    return a;
}

// Same but from f32 source (converted in-register).
__device__ __forceinline__ Frag load_frag_f32(const float* __restrict__ base, int hi) {
    Frag a;
    const float4* p0 = (const float4*)(base + 8 * hi);
    const float4* p1 = (const float4*)(base + 16 + 8 * hi);
    float4 x0 = p0[0], x1 = p0[1], y0 = p1[0], y1 = p1[1];
    a.s[0] = f2bf(x0.x);  a.s[1] = f2bf(x0.y);  a.s[2]  = f2bf(x0.z);  a.s[3]  = f2bf(x0.w);
    a.s[4] = f2bf(x1.x);  a.s[5] = f2bf(x1.y);  a.s[6]  = f2bf(x1.z);  a.s[7]  = f2bf(x1.w);
    a.s[8] = f2bf(y0.x);  a.s[9] = f2bf(y0.y);  a.s[10] = f2bf(y0.z);  a.s[11] = f2bf(y0.w);
    a.s[12] = f2bf(y1.x); a.s[13] = f2bf(y1.y); a.s[14] = f2bf(y1.z); a.s[15] = f2bf(y1.w);
    return a;
}

// ---------------------------------------------------------------------------
// Prep kernels
// ---------------------------------------------------------------------------

__global__ __launch_bounds__(256) void egnn_h2bf_kernel(
    const float* __restrict__ h, unsigned short* __restrict__ hbf, int n)
{
    int i = blockIdx.x * blockDim.x + threadIdx.x;
    if (i < n) hbf[i] = f2bf(h[i]);
}

// tail[e][0..63] bf16 = [radial, edge_attr[0..31], zeros]  (K chunks 8,9 of edge MLP input)
__global__ __launch_bounds__(256) void egnn_tail_kernel(
    const float* __restrict__ coords, const float* __restrict__ edge_attr,
    const int* __restrict__ row, const int* __restrict__ col,
    unsigned short* __restrict__ tail)
{
    int e = blockIdx.x * blockDim.x + threadIdx.x;
    if (e >= N_EDGES) return;
    int r = row[e], c = col[e];
    float dx = coords[r * 3 + 0] - coords[c * 3 + 0];
    float dy = coords[r * 3 + 1] - coords[c * 3 + 1];
    float dz = coords[r * 3 + 2] - coords[c * 3 + 2];
    float radial = dx * dx + dy * dy + dz * dz;
    unsigned short* t = tail + (size_t)e * 64;
    t[0] = f2bf(radial);
#pragma unroll
    for (int j = 0; j < 32; ++j) t[1 + j] = f2bf(edge_attr[(size_t)e * 32 + j]);
#pragma unroll
    for (int j = 33; j < 64; ++j) t[j] = 0;
}

// Pack f32 weight [K][128] into per-lane B fragments: frag(kt,nt,lane)[e] =
// W[kt*32 + 8*(lane>>4) + (e&7) + 16*(e>>3)][nt*16 + (lane&15)]  (zero-padded past Kvalid).
__global__ __launch_bounds__(256) void egnn_pack_w_kernel(
    const float* __restrict__ W, unsigned short* __restrict__ out, int Kvalid, int KT)
{
    int idx = blockIdx.x * blockDim.x + threadIdx.x;
    if (idx >= KT * NTILES * 32) return;
    int lane = idx & 31;
    int nt   = (idx >> 5) & 7;
    int kt   = idx >> 8;
    int n     = nt * 16 + (lane & 15);
    int kbase = kt * 32 + (lane >> 4) * 8;
    unsigned short* o = out + (size_t)idx * 16;
#pragma unroll
    for (int e = 0; e < 16; ++e) {
        int k = kbase + (e & 7) + 16 * (e >> 3);
        float v = (k < Kvalid) ? W[(size_t)k * HDIM + n] : 0.0f;
        o[e] = f2bf(v);
    }
}

__global__ __launch_bounds__(256) void egnn_zero_kernel(float* __restrict__ p, int n4) {
    int i = blockIdx.x * blockDim.x + threadIdx.x;
    if (i < n4) ((float4*)p)[i] = make_float4(0.f, 0.f, 0.f, 0.f);
}

// ---------------------------------------------------------------------------
// Edge MLP (fused gather -> MLP1 -> MLP2 -> scatter-add).
// Each wave owns TWO 16-edge M-tiles and reuses every B fragment for both
// (halves weight-feed bytes per WMMA). Stage-2 weights (32KB) are staged into
// LDS once per block via async-to-LDS, overlapped with stage-1 compute.
// Stage-2 runs per tile sequentially, reusing one LDS transpose slab.
// ---------------------------------------------------------------------------
__global__ __launch_bounds__(128) void egnn_edge_kernel(
    const unsigned short* __restrict__ hbf,
    const int* __restrict__ row, const int* __restrict__ col,
    const unsigned short* __restrict__ tail,
    const unsigned short* __restrict__ w1p, const float* __restrict__ b1,
    const unsigned short* __restrict__ w2p, const float* __restrict__ b2,
    float* __restrict__ agg)
{
    __shared__ __align__(32) unsigned short lds_m1[4][16 * PAD];
    __shared__ __align__(32) unsigned short w2lds[KT_E2 * NTILES * 32 * 16]; // 32KB

    const int wave = threadIdx.x >> 5;
    const int lane = threadIdx.x & 31;
    const int tile = blockIdx.x * 4 + wave;          // grid divides exactly
    const int e0 = tile * 32;                        // 32 edges per wave
    const int m  = lane & 15;        // A row / B,D column within tile
    const int hi = lane >> 4;
    const int em0 = e0 + m;
    const int em1 = e0 + 16 + m;
    const int rm0 = row[em0], cm0 = col[em0];
    const int rm1 = row[em1], cm1 = col[em1];

    // ---- kick off the stage-2 weight staging (16B packets, 16 per thread)
#pragma unroll
    for (int i = 0; i < 16; ++i) {
        int pkt = threadIdx.x + i * 128;
#if __has_builtin(__builtin_amdgcn_global_load_async_to_lds_b128)
        __builtin_amdgcn_global_load_async_to_lds_b128(
            (GLOBAL_AS v4i*)(w2p + pkt * 8),
            (LDS_AS v4i*)(w2lds + pkt * 8), 0, 0);
#else
        *(v8us*)(w2lds + pkt * 8) = *(const v8us*)(w2p + pkt * 8);
#endif
    }

    // prefetch the next dual-tile's tail block while we compute
    if (em0 + 128 < N_EDGES)
        __builtin_prefetch(tail + (size_t)(em0 + 128) * 64, 0, 1);

    const v16us* w1v = (const v16us*)w1p;

    // ---- stage 1: m1 = silu([h_r | h_c | radial | ea | 0] @ W1 + b1), K = 320
    v8f accA[NTILES], accB[NTILES];
#pragma unroll
    for (int nt = 0; nt < NTILES; ++nt) {
        accA[nt] = (v8f){0, 0, 0, 0, 0, 0, 0, 0};
        accB[nt] = (v8f){0, 0, 0, 0, 0, 0, 0, 0};
    }

#pragma unroll
    for (int kt = 0; kt < KT_E1; ++kt) {
        Frag aA, aB;
        if (kt < 4) {
            aA = load_frag_bf16(hbf + (size_t)rm0 * HDIM + kt * 32, hi);
            aB = load_frag_bf16(hbf + (size_t)rm1 * HDIM + kt * 32, hi);
        } else if (kt < 8) {
            aA = load_frag_bf16(hbf + (size_t)cm0 * HDIM + (kt - 4) * 32, hi);
            aB = load_frag_bf16(hbf + (size_t)cm1 * HDIM + (kt - 4) * 32, hi);
        } else {
            aA = load_frag_bf16(tail + (size_t)em0 * 64 + (kt - 8) * 32, hi);
            aB = load_frag_bf16(tail + (size_t)em1 * 64 + (kt - 8) * 32, hi);
        }
        Frag b[NTILES];
#pragma unroll
        for (int nt = 0; nt < NTILES; ++nt)
            b[nt].us = w1v[(kt * NTILES + nt) * 32 + lane];
#pragma unroll
        for (int nt = 0; nt < NTILES; ++nt) {
            accA[nt] = wmma_bf16(aA, b[nt], accA[nt]);
            accB[nt] = wmma_bf16(aB, b[nt], accB[nt]);
        }
    }

    // make the staged stage-2 weights visible to every wave
#if __has_builtin(__builtin_amdgcn_global_load_async_to_lds_b128)
#if __has_builtin(__builtin_amdgcn_s_wait_asynccnt)
    __builtin_amdgcn_s_wait_asynccnt(0);
#else
    asm volatile("s_wait_asynccnt 0" ::: "memory");
#endif
#endif
    __syncthreads();

    unsigned short* my = &lds_m1[wave][0];

    // ---- stage 2 + scatter, per tile (slab reused; same-wave DS is in-order)
#pragma unroll
    for (int t = 0; t < 2; ++t) {
        const v8f* acc = t ? accB : accA;
        const int ebase = e0 + t * 16;

        // D layout -> LDS in A layout
#pragma unroll
        for (int nt = 0; nt < NTILES; ++nt) {
            float bb = b1[nt * 16 + m];
#pragma unroll
            for (int r = 0; r < 8; ++r) {
                float v = silu_f(acc[nt][r] + bb);
                my[(r + 8 * hi) * PAD + nt * 16 + m] = f2bf(v);
            }
        }

        v8f acc2[NTILES];
#pragma unroll
        for (int nt = 0; nt < NTILES; ++nt) acc2[nt] = (v8f){0, 0, 0, 0, 0, 0, 0, 0};

#pragma unroll
        for (int kt = 0; kt < KT_E2; ++kt) {
            Frag a = load_frag_bf16(my + m * PAD + kt * 32, hi);
            Frag b[NTILES];
#pragma unroll
            for (int nt = 0; nt < NTILES; ++nt)
                b[nt].us = *(const v16us*)(w2lds + ((kt * NTILES + nt) * 32 + lane) * 16);
#pragma unroll
            for (int nt = 0; nt < NTILES; ++nt)
                acc2[nt] = wmma_bf16(a, b[nt], acc2[nt]);
        }

        int ri[8];
#pragma unroll
        for (int r = 0; r < 8; ++r) ri[r] = row[ebase + r + 8 * hi];
#pragma unroll
        for (int nt = 0; nt < NTILES; ++nt) {
            float bb = b2[nt * 16 + m];
#pragma unroll
            for (int r = 0; r < 8; ++r) {
                float v = silu_f(acc2[nt][r] + bb);
                atomicAdd(agg + (size_t)ri[r] * HDIM + nt * 16 + m, v);
            }
        }
    }
}

// ---------------------------------------------------------------------------
// Node MLP: out = silu([h|agg] @ Nw1 + nb1) @ Nw2 + nb2 ; h = out (+h if rec)
// Writes both f32 master h and the bf16 mirror used by the next edge pass.
// ---------------------------------------------------------------------------
__global__ __launch_bounds__(128) void egnn_node_kernel(
    const float* __restrict__ hin, const unsigned short* __restrict__ hbf_in,
    const float* __restrict__ agg,
    const unsigned short* __restrict__ w1p, const float* __restrict__ b1,
    const unsigned short* __restrict__ w2p, const float* __restrict__ b2,
    float* __restrict__ hout, unsigned short* __restrict__ hbf_out, int recurrent)
{
    __shared__ __align__(32) unsigned short lds_t[4][16 * PAD];
    const int wave = threadIdx.x >> 5;
    const int lane = threadIdx.x & 31;
    const int tile = blockIdx.x * 4 + wave;
    if (tile >= N_NODES / 16) return;
    const int n0 = tile * 16;
    const int m  = lane & 15;
    const int hi = lane >> 4;
    const int nd = n0 + m;

    const v16us* w1v = (const v16us*)w1p;
    const v16us* w2v = (const v16us*)w2p;

    v8f acc[NTILES];
#pragma unroll
    for (int nt = 0; nt < NTILES; ++nt) acc[nt] = (v8f){0, 0, 0, 0, 0, 0, 0, 0};

#pragma unroll
    for (int kt = 0; kt < KT_N1; ++kt) {
        Frag a;
        if (kt < 4) a = load_frag_bf16(hbf_in + (size_t)nd * HDIM + kt * 32, hi);
        else        a = load_frag_f32(agg + (size_t)nd * HDIM + (kt - 4) * 32, hi);
        Frag b[NTILES];
#pragma unroll
        for (int nt = 0; nt < NTILES; ++nt)
            b[nt].us = w1v[(kt * NTILES + nt) * 32 + lane];
#pragma unroll
        for (int nt = 0; nt < NTILES; ++nt)
            acc[nt] = wmma_bf16(a, b[nt], acc[nt]);
    }

    unsigned short* my = &lds_t[wave][0];
#pragma unroll
    for (int nt = 0; nt < NTILES; ++nt) {
        float bb = b1[nt * 16 + m];
#pragma unroll
        for (int r = 0; r < 8; ++r) {
            float v = silu_f(acc[nt][r] + bb);
            my[(r + 8 * hi) * PAD + nt * 16 + m] = f2bf(v);
        }
    }

    v8f acc2[NTILES];
#pragma unroll
    for (int nt = 0; nt < NTILES; ++nt) acc2[nt] = (v8f){0, 0, 0, 0, 0, 0, 0, 0};

#pragma unroll
    for (int kt = 0; kt < KT_N2; ++kt) {
        Frag a = load_frag_bf16(my + m * PAD + kt * 32, hi);
        Frag b[NTILES];
#pragma unroll
        for (int nt = 0; nt < NTILES; ++nt)
            b[nt].us = w2v[(kt * NTILES + nt) * 32 + lane];
#pragma unroll
        for (int nt = 0; nt < NTILES; ++nt)
            acc2[nt] = wmma_bf16(a, b[nt], acc2[nt]);
    }

#pragma unroll
    for (int nt = 0; nt < NTILES; ++nt) {
        float bb = b2[nt * 16 + m];
#pragma unroll
        for (int r = 0; r < 8; ++r) {
            int node = n0 + r + 8 * hi;
            size_t idx = (size_t)node * HDIM + nt * 16 + m;
            float v = acc2[nt][r] + bb;
            if (recurrent) v += hin[idx];
            hout[idx] = v;
            hbf_out[idx] = f2bf(v);
        }
    }
}

// ---------------------------------------------------------------------------
// LayerNorm over H=128 : one wave per node
// ---------------------------------------------------------------------------
__global__ __launch_bounds__(256) void egnn_ln_kernel(
    const float* __restrict__ h, const float* __restrict__ g,
    const float* __restrict__ b, float* __restrict__ out)
{
    const int wave = threadIdx.x >> 5;
    const int lane = threadIdx.x & 31;
    const int node = blockIdx.x * 8 + wave;
    if (node >= N_NODES) return;
    float4 v = ((const float4*)(h + (size_t)node * HDIM))[lane];
    float s = v.x + v.y + v.z + v.w;
#pragma unroll
    for (int off = 16; off > 0; off >>= 1) s += __shfl_xor(s, off, 32);
    float mu = s * (1.0f / HDIM);
    float dx = v.x - mu, dy = v.y - mu, dz = v.z - mu, dw = v.w - mu;
    float q = dx * dx + dy * dy + dz * dz + dw * dw;
#pragma unroll
    for (int off = 16; off > 0; off >>= 1) q += __shfl_xor(q, off, 32);
    float rstd = rsqrtf(q * (1.0f / HDIM) + 1e-5f);
    float4 gv = ((const float4*)g)[lane];
    float4 bv = ((const float4*)b)[lane];
    float4 o;
    o.x = dx * rstd * gv.x + bv.x;
    o.y = dy * rstd * gv.y + bv.y;
    o.z = dz * rstd * gv.z + bv.z;
    o.w = dw * rstd * gv.w + bv.w;
    ((float4*)(out + (size_t)node * HDIM))[lane] = o;
}

// ---------------------------------------------------------------------------
// Host launcher
// ---------------------------------------------------------------------------
extern "C" void kernel_launch(void* const* d_in, const int* in_sizes, int n_in,
                              void* d_out, int out_size, void* d_ws, size_t ws_size,
                              hipStream_t stream)
{
    (void)in_sizes; (void)n_in; (void)out_size; (void)ws_size;

    const float* h0        = (const float*)d_in[0];
    const float* coords    = (const float*)d_in[1];
    const float* edge_attr = (const float*)d_in[2];
    const int*   edges     = (const int*)d_in[3];
    const float* ew1 = (const float*)d_in[4];
    const float* eb1 = (const float*)d_in[5];
    const float* ew2 = (const float*)d_in[6];
    const float* eb2 = (const float*)d_in[7];
    const float* nw1 = (const float*)d_in[8];
    const float* nb1 = (const float*)d_in[9];
    const float* nw2 = (const float*)d_in[10];
    const float* nb2 = (const float*)d_in[11];
    const float* ln_g = (const float*)d_in[12];
    const float* ln_b = (const float*)d_in[13];

    const int* row = edges;
    const int* col = edges + N_EDGES;

    // workspace carve-up (256B aligned)
    char* ws = (char*)d_ws;
    size_t off = 0;
    auto carve = [&](size_t bytes) {
        char* p = ws + off;
        off = (off + bytes + 255) & ~(size_t)255;
        return (void*)p;
    };
    unsigned short* tail = (unsigned short*)carve((size_t)N_EDGES * 64 * 2);
    float* agg  = (float*)carve((size_t)N_NODES * HDIM * 4);
    float* hcur = (float*)carve((size_t)N_NODES * HDIM * 4);
    unsigned short* hbf = (unsigned short*)carve((size_t)N_NODES * HDIM * 2);
    const size_t FR_E1 = (size_t)KT_E1 * NTILES * 32 * 16;   // ushorts / layer
    const size_t FR_E2 = (size_t)KT_E2 * NTILES * 32 * 16;
    const size_t FR_N1 = (size_t)KT_N1 * NTILES * 32 * 16;
    const size_t FR_N2 = (size_t)KT_N2 * NTILES * 32 * 16;
    unsigned short* w1p = (unsigned short*)carve(NLAYERS * FR_E1 * 2);
    unsigned short* w2p = (unsigned short*)carve(NLAYERS * FR_E2 * 2);
    unsigned short* n1p = (unsigned short*)carve(NLAYERS * FR_N1 * 2);
    unsigned short* n2p = (unsigned short*)carve(NLAYERS * FR_N2 * 2);

    // ---- prep: bf16 h mirror, per-edge tail features, weight fragment packing
    egnn_h2bf_kernel<<<(N_NODES * HDIM + 255) / 256, 256, 0, stream>>>(
        h0, hbf, N_NODES * HDIM);
    egnn_tail_kernel<<<(N_EDGES + 255) / 256, 256, 0, stream>>>(coords, edge_attr, row, col, tail);
    for (int l = 0; l < NLAYERS; ++l) {
        egnn_pack_w_kernel<<<(KT_E1 * NTILES * 32 + 255) / 256, 256, 0, stream>>>(
            ew1 + (size_t)l * KE1_VALID * HDIM, w1p + l * FR_E1, KE1_VALID, KT_E1);
        egnn_pack_w_kernel<<<(KT_E2 * NTILES * 32 + 255) / 256, 256, 0, stream>>>(
            ew2 + (size_t)l * HDIM * HDIM, w2p + l * FR_E2, HDIM, KT_E2);
        egnn_pack_w_kernel<<<(KT_N1 * NTILES * 32 + 255) / 256, 256, 0, stream>>>(
            nw1 + (size_t)l * 2 * HDIM * HDIM, n1p + l * FR_N1, 2 * HDIM, KT_N1);
        egnn_pack_w_kernel<<<(KT_N2 * NTILES * 32 + 255) / 256, 256, 0, stream>>>(
            nw2 + (size_t)l * HDIM * HDIM, n2p + l * FR_N2, HDIM, KT_N2);
    }

    // ---- layers
    for (int l = 0; l < NLAYERS; ++l) {
        const float* hin = (l == 0) ? h0 : hcur;
        egnn_zero_kernel<<<(N_NODES * HDIM / 4 + 255) / 256, 256, 0, stream>>>(
            agg, N_NODES * HDIM / 4);
        egnn_edge_kernel<<<N_EDGES / 32 / 4, 128, 0, stream>>>(
            hbf, row, col, tail,
            w1p + l * FR_E1, eb1 + (size_t)l * HDIM,
            w2p + l * FR_E2, eb2 + (size_t)l * HDIM, agg);
        egnn_node_kernel<<<(N_NODES / 16 + 3) / 4, 128, 0, stream>>>(
            hin, hbf, agg,
            n1p + l * FR_N1, nb1 + (size_t)l * HDIM,
            n2p + l * FR_N2, nb2 + (size_t)l * HDIM,
            hcur, hbf, l > 0 ? 1 : 0);
    }

    // ---- final LayerNorm
    egnn_ln_kernel<<<(N_NODES + 7) / 8, 256, 0, stream>>>(hcur, ln_g, ln_b, (float*)d_out);
}
